// MoELayer_41128606827159
// MI455X (gfx1250) — matrix operand order repaired and checked by reading, not compiled, not run on hardware
//
#include <hip/hip_runtime.h>
#include <hip/hip_bf16.h>
#include <math.h>

typedef __bf16 bf16;
typedef bf16  v16bf __attribute__((ext_vector_type(16)));
typedef float v8f   __attribute__((ext_vector_type(8)));

#define D_MODEL 1024
#define D_FF    4096
#define NE      8
#define TOKENS  8192
#define FC      512                 // F chunk held in LDS per iteration
#define NCHUNK  (D_FF / FC)         // 8
#define XS_STRIDE (D_MODEL + 8)     // 1032 bf16 elems (2064 B, 16B aligned)
#define HC_STRIDE (FC + 8)          // 520 bf16 elems  (1040 B, 16B aligned)

// ---------------- init: zero output + expert counters ----------------
__global__ void k_init(float* __restrict__ out, int outN, int* __restrict__ cnt) {
    int i = blockIdx.x * blockDim.x + threadIdx.x;
    int stride = gridDim.x * blockDim.x;
    for (int j = i; j < outN; j += stride) out[j] = 0.0f;
    if (i < NE) cnt[i] = 0;
}

// ---------------- fp32 -> bf16 copy of x (coalesced) ----------------
__global__ void k_cvt(const float* __restrict__ x, bf16* __restrict__ xb, int n) {
    int i = blockIdx.x * blockDim.x + threadIdx.x;
    int stride = gridDim.x * blockDim.x;
    for (int j = i; j < n; j += stride) xb[j] = (bf16)x[j];
}

// ------ transpose + convert: W [R,C] f32 row-major -> WT [C,R] bf16 row-major ------
// one expert per blockIdx.z; 32x32 tiles via LDS, coalesced read and write
__global__ __launch_bounds__(256)
void k_tr(const float* __restrict__ W, bf16* __restrict__ WT, int R, int C) {
    __shared__ float t[32][33];
    const float* We = W  + (size_t)blockIdx.z * R * C;
    bf16*       WTe = WT + (size_t)blockIdx.z * R * C;
    int gx = blockIdx.x * 32, gy = blockIdx.y * 32;
#pragma unroll
    for (int j = 0; j < 32; j += 8)
        t[threadIdx.y + j][threadIdx.x] =
            We[(size_t)(gy + threadIdx.y + j) * C + gx + threadIdx.x];
    __syncthreads();
#pragma unroll
    for (int j = 0; j < 32; j += 8)
        WTe[(size_t)(gx + threadIdx.y + j) * R + gy + threadIdx.x] =
            (bf16)t[threadIdx.x][threadIdx.y + j];
}

// ---------------- gating: one wave32 per token ----------------
__global__ __launch_bounds__(256)
void k_gate(const float* __restrict__ x, const float* __restrict__ Wg,
            int* __restrict__ cnt, int* __restrict__ tokList, float* __restrict__ wList) {
    int wave = threadIdx.x >> 5;
    int lane = threadIdx.x & 31;
    int tok  = blockIdx.x * (blockDim.x >> 5) + wave;
    if (tok >= TOKENS) return;

    float acc[NE];
#pragma unroll
    for (int e = 0; e < NE; ++e) acc[e] = 0.0f;

    const float* xr = x + (size_t)tok * D_MODEL;
    for (int i = lane; i < D_MODEL; i += 32) {
        float xv = xr[i];
        const float* wr = Wg + (size_t)i * NE;
#pragma unroll
        for (int e = 0; e < NE; ++e) acc[e] += xv * wr[e];
    }
#pragma unroll
    for (int e = 0; e < NE; ++e) {
#pragma unroll
        for (int off = 16; off >= 1; off >>= 1)
            acc[e] += __shfl_down(acc[e], off, 32);
    }

    if (lane == 0) {
        float m = acc[0];
#pragma unroll
        for (int e = 1; e < NE; ++e) m = fmaxf(m, acc[e]);
        float p[NE]; float s = 0.0f;
#pragma unroll
        for (int e = 0; e < NE; ++e) { p[e] = __expf(acc[e] - m); s += p[e]; }
        float inv = 1.0f / s;
#pragma unroll
        for (int e = 0; e < NE; ++e) p[e] *= inv;

        int i1 = 0;
#pragma unroll
        for (int e = 1; e < NE; ++e) if (p[e] > p[i1]) i1 = e;
        int i2 = (i1 == 0) ? 1 : 0;
#pragma unroll
        for (int e = 0; e < NE; ++e) if (e != i1 && p[e] > p[i2]) i2 = e;

        float denom = p[i1] + p[i2] + 1e-8f;
        float w1 = p[i1] / denom, w2 = p[i2] / denom;

        int pos1 = atomicAdd(&cnt[i1], 1);
        tokList[i1 * TOKENS + pos1] = tok;  wList[i1 * TOKENS + pos1] = w1;
        int pos2 = atomicAdd(&cnt[i2], 1);
        tokList[i2 * TOKENS + pos2] = tok;  wList[i2 * TOKENS + pos2] = w2;
    }
}

// -------- A fragment (16x32 bf16, MxK) from LDS: two contiguous b128 runs --------
// lane (g = lane>>4, m = lane&15): elems 0..15 <-> K = kbase + 16g + 0..15
__device__ __forceinline__ v16bf wmma_a_from_lds(const bf16* row0, int stride,
                                                 int kbase, int lane) {
    int g = lane >> 4, m = lane & 15;
    const bf16* p = row0 + m * stride + kbase + 16 * g;
    union { uint4 q[2]; v16bf v; } u;
    u.q[0] = *(const uint4*)(p);
    u.q[1] = *(const uint4*)(p + 8);
    return u.v;
}

// -------- B fragment (32x16 bf16, KxN) from transposed bf16 weights --------
// WT is [N_total, K_total] row-major; lane holds col n = lane&15, K = k0+16g+0..15
// contiguous 32 bytes -> two global_load_b128, element order matches K=16g+2v,2v+1
__device__ __forceinline__ v16bf wmma_b_from_bf16T(const bf16* __restrict__ WT, int ld,
                                                   int k0, int col, int lane) {
    int g = lane >> 4, n = lane & 15;
    const bf16* p = WT + (size_t)(col + n) * ld + k0 + 16 * g;
    union { uint4 q[2]; v16bf v; } u;
    u.q[0] = *(const uint4*)(p);
    u.q[1] = *(const uint4*)(p + 8);
    return u.v;
}

// ---------------- fused expert FFN on gathered token tiles ----------------
__global__ __launch_bounds__(256)
void k_moe(const bf16* __restrict__ xb,
           const bf16* __restrict__ W1T, const bf16* __restrict__ W2T,
           const bf16* __restrict__ W3T,
           const int* __restrict__ cnt, const int* __restrict__ tokList,
           const float* __restrict__ wList,
           float* __restrict__ out) {
    __shared__ __align__(16) bf16  Xs[16 * XS_STRIDE];   // gathered X tile, bf16
    __shared__ __align__(16) bf16  Hc[16 * HC_STRIDE];   // silu(xW1)*xW3 chunk, bf16
    __shared__ int   s_tok[16];
    __shared__ float s_w[16];

    const int e    = blockIdx.y;
    const int n_e  = cnt[e];
    const int tile = blockIdx.x;
    if (tile * 16 >= n_e) return;          // block-uniform exit

    const int tid  = threadIdx.x;
    const int wv   = tid >> 5;             // wave 0..7
    const int lane = tid & 31;

    if (tid < 16) {
        int slot = tile * 16 + tid;
        if (slot < n_e) {
            s_tok[tid] = tokList[e * TOKENS + slot];
            s_w[tid]   = wList[e * TOKENS + slot];
        } else {
            s_tok[tid] = tokList[e * TOKENS];   // pad row, weight 0
            s_w[tid]   = 0.0f;
        }
    }
    __syncthreads();

    // Stage 16 gathered x rows into LDS (16-byte vector copies)
    for (int i = tid * 8; i < 16 * D_MODEL; i += 256 * 8) {
        int r = i >> 10;
        int c = i & (D_MODEL - 1);
        const bf16* src = xb + (size_t)s_tok[r] * D_MODEL + c;
        *(uint4*)&Xs[r * XS_STRIDE + c] = *(const uint4*)src;
    }
    __syncthreads();

    const bf16* W1e = W1T + (size_t)e * D_MODEL * D_FF;   // [F, D] bf16
    const bf16* W3e = W3T + (size_t)e * D_MODEL * D_FF;   // [F, D] bf16
    const bf16* W2e = W2T + (size_t)e * D_FF * D_MODEL;   // [D, F] bf16

    v8f zero = {};
    v8f accO[8];                            // 16 x 128 out cols per wave
#pragma unroll
    for (int t = 0; t < 8; ++t) accO[t] = zero;

    for (int fc = 0; fc < NCHUNK; ++fc) {
        // ---- Pass A: Hc[16,FC] = silu(X@W1) * (X@W3) over this F chunk ----
        v8f acc1[4], acc3[4];
#pragma unroll
        for (int t = 0; t < 4; ++t) { acc1[t] = zero; acc3[t] = zero; }

        for (int kt = 0; kt < D_MODEL / 32; ++kt) {
            v16bf a = wmma_a_from_lds(Xs, XS_STRIDE, kt * 32, lane);
#pragma unroll
            for (int t = 0; t < 4; ++t) {
                int colF = fc * FC + wv * 64 + t * 16;     // column in F of W1/W3
                v16bf b1 = wmma_b_from_bf16T(W1e, D_MODEL, kt * 32, colF, lane);
                acc1[t] = __builtin_amdgcn_wmma_f32_16x16x32_bf16(
                              false, a, false, b1, (short)0, acc1[t], false, false);
                v16bf b3 = wmma_b_from_bf16T(W3e, D_MODEL, kt * 32, colF, lane);
                acc3[t] = __builtin_amdgcn_wmma_f32_16x16x32_bf16(
                              false, a, false, b3, (short)0, acc3[t], false, false);
            }
        }
        {   // silu * gate -> LDS (bf16), C-layout: M = r + 8*(lane>>4), N = lane&15
            int g = lane >> 4, n = lane & 15;
#pragma unroll
            for (int t = 0; t < 4; ++t) {
                int cb = wv * 64 + t * 16 + n;
#pragma unroll
                for (int r = 0; r < 8; ++r) {
                    float h1 = acc1[t][r], h3 = acc3[t][r];
                    float sv = h1 / (1.0f + __expf(-h1));
                    Hc[(r + 8 * g) * HC_STRIDE + cb] = (bf16)(sv * h3);
                }
            }
        }
        __syncthreads();   // Hc fully written before pass B reads

        // ---- Pass B: accO += Hc[16,FC] @ W2 chunk ----
        for (int kt = 0; kt < FC / 32; ++kt) {
            v16bf a = wmma_a_from_lds(Hc, HC_STRIDE, kt * 32, lane);
#pragma unroll
            for (int t = 0; t < 8; ++t) {
                int col = wv * 128 + t * 16;               // column in D of out
                v16bf b = wmma_b_from_bf16T(W2e, D_FF, fc * FC + kt * 32, col, lane);
                accO[t] = __builtin_amdgcn_wmma_f32_16x16x32_bf16(
                              false, a, false, b, (short)0, accO[t], false, false);
            }
        }
        __syncthreads();   // done reading Hc before next chunk overwrites
    }

    // ---- weighted scatter: out[token] += w * row (exactly 2 adds/elem, commutative) ----
    {
        int g = lane >> 4, n = lane & 15;
#pragma unroll
        for (int t = 0; t < 8; ++t) {
            int col = wv * 128 + t * 16 + n;
#pragma unroll
            for (int r = 0; r < 8; ++r) {
                int M = r + 8 * g;
                float wgt = s_w[M];
                if (wgt != 0.0f) {
                    unsafeAtomicAdd(&out[(size_t)s_tok[M] * D_MODEL + col],
                                    wgt * accO[t][r]);
                }
            }
        }
    }
}

// ---------------- host entry ----------------
extern "C" void kernel_launch(void* const* d_in, const int* in_sizes, int n_in,
                              void* d_out, int out_size, void* d_ws, size_t ws_size,
                              hipStream_t stream) {
    const float* x  = (const float*)d_in[0];
    const float* Wg = (const float*)d_in[1];
    const float* W1 = (const float*)d_in[2];
    const float* W2 = (const float*)d_in[3];
    const float* W3 = (const float*)d_in[4];
    float* out = (float*)d_out;

    const size_t WSZ = (size_t)NE * D_MODEL * D_FF;   // elems per weight tensor

    char* ws = (char*)d_ws;
    size_t off = 0;
    int*   cnt  = (int*)(ws + off);            off += 1024;
    int*   tokL = (int*)(ws + off);            off += (size_t)NE * TOKENS * 4;
    float* wL   = (float*)(ws + off);          off += (size_t)NE * TOKENS * 4;
    bf16*  xbuf = (bf16*)(ws + off);           off += (size_t)TOKENS * D_MODEL * 2;
    bf16*  W1T  = (bf16*)(ws + off);           off += WSZ * 2;
    bf16*  W3T  = (bf16*)(ws + off);           off += WSZ * 2;
    bf16*  W2T  = (bf16*)(ws + off);           off += WSZ * 2;

    k_init<<<512, 256, 0, stream>>>(out, out_size, cnt);
    k_cvt <<<1024, 256, 0, stream>>>(x, xbuf, TOKENS * D_MODEL);

    dim3 trb(32, 8);
    k_tr<<<dim3(D_FF / 32, D_MODEL / 32, NE), trb, 0, stream>>>(W1, W1T, D_MODEL, D_FF);
    k_tr<<<dim3(D_FF / 32, D_MODEL / 32, NE), trb, 0, stream>>>(W3, W3T, D_MODEL, D_FF);
    k_tr<<<dim3(D_MODEL / 32, D_FF / 32, NE), trb, 0, stream>>>(W2, W2T, D_FF, D_MODEL);

    k_gate<<<TOKENS / 8, 256, 0, stream>>>(x, Wg, cnt, tokL, wL);

    dim3 grid(TOKENS / 16, NE);   // worst case: all tokens routed to one expert
    k_moe<<<grid, 256, 0, stream>>>(xbuf, W1T, W2T, W3T, cnt, tokL, wL, out);
}
